// PointNetSetAbstraction_grid_31997506355427
// MI455X (gfx1250) — matrix operand order repaired and checked by reading, not compiled, not run on hardware
//
#include <hip/hip_runtime.h>
#include <hip/hip_bf16.h>
#include <math.h>

// ---------------------------------------------------------------------------
// Problem constants (match reference)
// ---------------------------------------------------------------------------
#define BB      8
#define NN      4096
#define DD      64
#define SS      1024
#define KK      32
#define G3      125
#define C_IN    67
#define C_PAD   96          // channels padded to 3 WMMA K-steps / 6 16-runs
#define RADIUS  0.2f
#define INV_RXYZ (4.0f / 1.7320508075688772f)   // (G-1)/sqrt(3)
#define EPSF    1e-6f
#define NGRP    (BB*SS)     // 8192 groups
// conv0: K ordered tau-major, channels padded to 96 per tap: K = 27*96 = 2592
#define K0STEPS 81
// conv1: K ordered c-major, tau padded to 32: K = 64*32 = 2048
#define K1STEPS 64
#define OUT_XYZ_ELEMS (BB*3*SS)     // 24576

typedef __attribute__((ext_vector_type(16))) __bf16 v16bf;
typedef __attribute__((ext_vector_type(8)))  float  v8f;

#define WMMA_BF16(A,B,C) \
  __builtin_amdgcn_wmma_f32_16x16x32_bf16(false,(A),false,(B),(short)0,(C),false,false)

#define V8F_ZERO {0.f,0.f,0.f,0.f,0.f,0.f,0.f,0.f}

// ---------------------------------------------------------------------------
// Kernel 1: gather neighbors -> bf16 feature tile [grp][32 pts][96 ch]
// (point-major so MLP B-fragments are contiguous; cols 67..95 zero),
// fp32 locs [grp][32][3], and new_xyz output [B,3,S].
// ---------------------------------------------------------------------------
__global__ __launch_bounds__(128) void k_gather(
    const float* __restrict__ xyz, const float* __restrict__ points,
    const int* __restrict__ fps_idx, const int* __restrict__ ball_idx,
    float* __restrict__ out, __bf16* __restrict__ feat, float* __restrict__ locs)
{
  int grp = blockIdx.x;            // (b,s)
  int b = grp >> 10, s = grp & 1023;
  int t = threadIdx.x;             // 0..127
  int fidx = fps_idx[grp];

  if (t < 3)
    out[(b*3 + t)*SS + s] = xyz[(b*3 + t)*NN + fidx];

  int k  = t & 31;
  int cg = t >> 5;                 // 0..3
  int bidx = ball_idx[grp*KK + k];
  __bf16* f = feat + (size_t)grp * KK * C_PAD;

  if (cg == 0) {
    #pragma unroll
    for (int d = 0; d < 3; ++d) {
      float nx = xyz[(b*3 + d)*NN + fidx];
      float gx = (xyz[(b*3 + d)*NN + bidx] - nx) * (1.0f / RADIUS);
      locs[((size_t)grp*KK + k)*3 + d] = gx;
      f[k*C_PAD + d] = (__bf16)gx;
    }
  }
  #pragma unroll
  for (int i = 0; i < 16; ++i) {
    int c = 3 + cg*16 + i;          // 3..66
    f[k*C_PAD + c] = (__bf16)points[((size_t)b*DD + (c-3))*NN + bidx];
  }
  // zero-pad channels 67..95 (WMMA K padding)
  for (int idx = t; idx < (C_PAD - C_IN)*KK; idx += 128)
    f[(idx & 31)*C_PAD + C_IN + idx/KK] = (__bf16)0.0f;
}

// ---------------------------------------------------------------------------
// Kernel 2: pack fp32 weights into WMMA A-fragment order (bf16, 16x16x32).
// dst[((step*Mtiles+mt)*32+lane)*16+e] = A[mt*16+(lane&15)][kpat], where
// kpat = 32*step + 16*(lane>>4) + e.
// K-order modes:
//   0: direct, column k, zero for k >= p0 (p0 = Creal)
//   1: conv0  tau-major: c = k%96, tau = k/96 ; src W[m*1809 + c*27 + tau], c<67
//   2: conv1  c-major  : c = k/32, tau = k%32 ; src W[m*1728 + c*27 + tau], tau<27
// ---------------------------------------------------------------------------
__global__ void k_packA(const float* __restrict__ W, int mode, int Mtiles,
                        int Ksteps, int p0, __bf16* __restrict__ dst)
{
  int tid = blockIdx.x*256 + threadIdx.x;
  int total = Ksteps * Mtiles * 512;
  if (tid >= total) return;
  int e    = tid & 15;
  int lane = (tid >> 4) & 31;
  int mt   = (tid >> 9) % Mtiles;
  int step = tid / (512 * Mtiles);
  int m  = mt*16 + (lane & 15);
  int kh = lane >> 4;
  int v  = e >> 1, j = e & 1;
  int kk = step*32 + 16*kh + e;

  float val = 0.0f;
  if (mode == 0) {
    if (kk < p0) val = W[(size_t)m*p0 + kk];
  } else if (mode == 1) {
    int c = kk % 96, tau = kk / 96;
    if (c < C_IN) val = W[(size_t)m*1809 + c*27 + tau];
  } else {
    int c = kk >> 5, tau = kk & 31;
    if (tau < 27) val = W[(size_t)m*1728 + c*27 + tau];
  }
  dst[tid] = (__bf16)val;
}

// ---------------------------------------------------------------------------
// Kernel 3: fused grid-render + conv3d stack.  1 group / block (128 thr =
// 4 waves; wave = one 16-row M-tile).
// conv0: WMMA, grid tile transposed gfT[125][96] so B frags are contiguous.
// conv1: WMMA, N=1 (lane col 0), B = linear h0 buffer (contiguous frags).
// conv2: VALU (8K MAC).  Writes nf3d into out[24576 + (b*128+o)*1024 + s].
// ---------------------------------------------------------------------------
__global__ __launch_bounds__(128) void k_conv(
    const __bf16* __restrict__ feat, const float* __restrict__ locs,
    const __bf16* __restrict__ W0p, const __bf16* __restrict__ W1p,
    const float* __restrict__ b0, const float* __restrict__ s0, const float* __restrict__ t0,
    const float* __restrict__ b1, const float* __restrict__ s1, const float* __restrict__ t1,
    const float* __restrict__ W2, const float* __restrict__ b2,
    const float* __restrict__ s2, const float* __restrict__ t2,
    float* __restrict__ out)
{
  __shared__ unsigned short tofftab[32];     // tau -> td*25+th*5+tw
  __shared__ __bf16 ldsFeat[KK*C_PAD];       // [32 pts][96 ch]
  __shared__ __bf16 ldsGFT[G3*C_PAD];        // [125 cells][96 ch], transposed tile
  __shared__ float  ldsLoc[KK*3];
  __shared__ __bf16 ldsH0[64*32];            // conv0 out [64 ch][32 pos pad] == conv1 K buf
  __shared__ float  ldsP[64];
  __shared__ float  ldsH1[64];

  int t    = threadIdx.x;          // 0..127
  int lane = t & 31;
  int wv   = t >> 5;               // M-tile 0..3
  int grp  = blockIdx.x;

  if (t < 27) {
    int td = t/9, th = (t%9)/3, tw = t%3;
    tofftab[t] = (unsigned short)(td*25 + th*5 + tw);
  }
  // stage feat + locs, zero grid tile
  {
    const unsigned int* fsrc = (const unsigned int*)(feat + (size_t)grp*KK*C_PAD);
    unsigned int* fdst = (unsigned int*)ldsFeat;
    for (int i = t; i < KK*C_PAD/2; i += 128) fdst[i] = fsrc[i];
    const float* lsrc = locs + (size_t)grp*KK*3;
    for (int i = t; i < KK*3; i += 128) ldsLoc[i] = lsrc[i];
    unsigned int* gz = (unsigned int*)ldsGFT;
    for (int i = t; i < G3*C_PAD/2; i += 128) gz[i] = 0u;
  }
  __syncthreads();

  // --- grid render: one grid cell per thread (125 of 128 active) ---
  if (t < G3) {
    int g = t;
    float cx = -1.0f + 0.5f*(float)(g/25);
    float cy = -1.0f + 0.5f*(float)((g/5)%5);
    float cz = -1.0f + 0.5f*(float)(g%5);
    float wk[KK];
    float sum = 0.0f;
    #pragma unroll
    for (int k = 0; k < KK; ++k) {
      float dx = ldsLoc[k*3+0] - cx;
      float dy = ldsLoc[k*3+1] - cy;
      float dz = ldsLoc[k*3+2] - cz;
      float d  = sqrtf(dx*dx + dy*dy + dz*dz);
      float w  = 1.0f - d*INV_RXYZ;
      w = (w > 0.0f) ? w : 0.0f;
      wk[k] = w; sum += w;
    }
    float inv = 1.0f / (sum + EPSF);
    for (int c = 0; c < C_IN; ++c) {
      float acc = 0.0f;
      #pragma unroll
      for (int k = 0; k < KK; ++k) acc += (float)ldsFeat[k*C_PAD + c] * wk[k];
      ldsGFT[g*C_PAD + c] = (__bf16)(acc * inv);
    }
  }
  __syncthreads();

  // --- conv0 via WMMA: K = (tau major, 96-ch pad), contiguous B frags ---
  int kh = lane >> 4;
  int nl = lane & 15;
  int n0 = nl, n1 = 16 + nl;
  int gb0 = (n0/9)*25 + ((n0%9)/3)*5 + (n0%3);
  int gb1 = (n1 < 27) ? ((n1/9)*25 + ((n1%9)/3)*5 + (n1%3)) : 0;

  v8f acc0 = V8F_ZERO, acc1 = V8F_ZERO;
  for (int step = 0; step < K0STEPS; ++step) {
    v16bf a = *((const v16bf*)W0p + ((size_t)step*4 + wv)*32 + lane);
    int seg  = step*2 + kh;            // aligned 16-run index; 96/16=6 runs per tau
    int tau  = seg / 6;
    int c0   = (seg % 6) * 16;
    int toff = (int)tofftab[tau];
    v16bf bv0 = *(const v16bf*)&ldsGFT[(gb0 + toff)*C_PAD + c0];
    v16bf bv1 = *(const v16bf*)&ldsGFT[(gb1 + toff)*C_PAD + c0];
    acc0 = WMMA_BF16(a, bv0, acc0);
    acc1 = WMMA_BF16(a, bv1, acc1);
  }

  // --- BN + ReLU -> h0 [64 ch][32 pos] (cols >= 27 zeroed) ---
  #pragma unroll
  for (int v = 0; v < 8; ++v) {
    int m = wv*16 + v + 8*kh;
    float bbv = b0[m], ssv = s0[m], ttv = t0[m];
    float y0 = ssv*(acc0[v] + bbv) + ttv;  y0 = (y0 > 0.f) ? y0 : 0.f;
    ldsH0[m*32 + n0] = (__bf16)y0;
    float y1 = 0.0f;
    if (n1 < 27) {
      y1 = ssv*(acc1[v] + bbv) + ttv;  y1 = (y1 > 0.f) ? y1 : 0.f;
    }
    ldsH0[m*32 + n1] = (__bf16)y1;
  }
  __syncthreads();

  // --- conv1 via WMMA: K = 64*32 linear over h0, N=1 (col 0 only) ---
  {
    v16bf bz;
    #pragma unroll
    for (int e = 0; e < 16; ++e) bz[e] = (__bf16)0.0f;
    v8f acc = V8F_ZERO;
    for (int step = 0; step < K1STEPS; ++step) {
      v16bf a = *((const v16bf*)W1p + ((size_t)step*4 + wv)*32 + lane);
      v16bf b = bz;
      if (nl == 0)
        b = *(const v16bf*)&ldsH0[step*32 + 16*kh];
      acc = WMMA_BF16(a, b, acc);
    }
    if (nl == 0) {
      #pragma unroll
      for (int v = 0; v < 8; ++v)
        ldsP[wv*16 + v + 8*kh] = acc[v];
    }
  }
  __syncthreads();
  if (t < 64) {
    int o = t;
    float y = s1[o]*(ldsP[o] + b1[o]) + t1[o];  y = (y > 0.f) ? y : 0.f;
    ldsH1[o] = y;
  }
  __syncthreads();

  // --- conv2 (128x64 1x1) + write nf3d ---
  {
    int o2 = t;   // 0..127
    float acc = 0.0f;
    #pragma unroll 8
    for (int c = 0; c < 64; ++c) acc += W2[(size_t)o2*64 + c] * ldsH1[c];
    float y = s2[o2]*(acc + b2[o2]) + t2[o2];  y = (y > 0.f) ? y : 0.f;
    int b = grp >> 10, s = grp & 1023;
    out[OUT_XYZ_ELEMS + ((size_t)b*128 + o2)*SS + s] = y;
  }
}

// ---------------------------------------------------------------------------
// Kernel 4: point MLP branch.  1 group / block (128 thr = 4 waves).
// All activations stored point-major so B fragments are contiguous.
// ---------------------------------------------------------------------------
__global__ __launch_bounds__(128) void k_mlp(
    const __bf16* __restrict__ feat,
    const __bf16* __restrict__ Wap, const float* __restrict__ ba,
    const float* __restrict__ sa,   const float* __restrict__ ta,
    const __bf16* __restrict__ Wbp, const float* __restrict__ bbp,
    const float* __restrict__ sb,   const float* __restrict__ tb,
    const __bf16* __restrict__ Wcp, const float* __restrict__ bc,
    const float* __restrict__ sc,   const float* __restrict__ tc,
    float* __restrict__ out)
{
  __shared__ __bf16 ldsF [KK*C_PAD];   // [32 pts][96 ch]
  __shared__ __bf16 ldsY1[KK*64];      // [32 pts][64 ch]
  __shared__ __bf16 ldsY2[KK*64];      // [32 pts][64 ch]
  __shared__ float  ldsY3[128*32];     // [128 ch][32 pts]

  int t = threadIdx.x, lane = t & 31, wv = t >> 5;
  int grp = blockIdx.x;
  int kh = lane >> 4, nl = lane & 15;

  {
    const unsigned int* fsrc = (const unsigned int*)(feat + (size_t)grp*KK*C_PAD);
    unsigned int* fdst = (unsigned int*)ldsF;
    for (int i = t; i < KK*C_PAD/2; i += 128) fdst[i] = fsrc[i];
  }
  __syncthreads();

  // ---- layer A: [64,96] x [96,32] ----
  {
    v8f a0 = V8F_ZERO, a1 = V8F_ZERO;
    for (int step = 0; step < 3; ++step) {
      v16bf a = *((const v16bf*)Wap + ((size_t)step*4 + wv)*32 + lane);
      int kb = step*32 + 16*kh;
      v16bf b0v = *(const v16bf*)&ldsF[nl*C_PAD + kb];
      v16bf b1v = *(const v16bf*)&ldsF[(16 + nl)*C_PAD + kb];
      a0 = WMMA_BF16(a, b0v, a0);
      a1 = WMMA_BF16(a, b1v, a1);
    }
    #pragma unroll
    for (int v = 0; v < 8; ++v) {
      int m = wv*16 + v + 8*kh;
      float bv = ba[m], sv = sa[m], tv = ta[m];
      float y0 = sv*(a0[v] + bv) + tv;  y0 = (y0 > 0.f) ? y0 : 0.f;
      float y1 = sv*(a1[v] + bv) + tv;  y1 = (y1 > 0.f) ? y1 : 0.f;
      ldsY1[nl*64 + m]        = (__bf16)y0;
      ldsY1[(16 + nl)*64 + m] = (__bf16)y1;
    }
  }
  __syncthreads();

  // ---- layer B: [64,64] x [64,32] ----
  {
    v8f a0 = V8F_ZERO, a1 = V8F_ZERO;
    for (int step = 0; step < 2; ++step) {
      v16bf a = *((const v16bf*)Wbp + ((size_t)step*4 + wv)*32 + lane);
      int kb = step*32 + 16*kh;
      v16bf b0v = *(const v16bf*)&ldsY1[nl*64 + kb];
      v16bf b1v = *(const v16bf*)&ldsY1[(16 + nl)*64 + kb];
      a0 = WMMA_BF16(a, b0v, a0);
      a1 = WMMA_BF16(a, b1v, a1);
    }
    #pragma unroll
    for (int v = 0; v < 8; ++v) {
      int m = wv*16 + v + 8*kh;
      float bv = bbp[m], sv = sb[m], tv = tb[m];
      float y0 = sv*(a0[v] + bv) + tv;  y0 = (y0 > 0.f) ? y0 : 0.f;
      float y1 = sv*(a1[v] + bv) + tv;  y1 = (y1 > 0.f) ? y1 : 0.f;
      ldsY2[nl*64 + m]        = (__bf16)y0;
      ldsY2[(16 + nl)*64 + m] = (__bf16)y1;
    }
  }
  __syncthreads();

  // ---- layer C: [128,64] x [64,32], 2 M-tiles per wave ----
  #pragma unroll
  for (int mi = 0; mi < 2; ++mi) {
    int mt = wv*2 + mi;
    v8f a0 = V8F_ZERO, a1 = V8F_ZERO;
    for (int step = 0; step < 2; ++step) {
      v16bf a = *((const v16bf*)Wcp + ((size_t)step*8 + mt)*32 + lane);
      int kb = step*32 + 16*kh;
      v16bf b0v = *(const v16bf*)&ldsY2[nl*64 + kb];
      v16bf b1v = *(const v16bf*)&ldsY2[(16 + nl)*64 + kb];
      a0 = WMMA_BF16(a, b0v, a0);
      a1 = WMMA_BF16(a, b1v, a1);
    }
    #pragma unroll
    for (int v = 0; v < 8; ++v) {
      int m = mt*16 + v + 8*kh;
      float bv = bc[m], sv = sc[m], tv = tc[m];
      float y0 = sv*(a0[v] + bv) + tv;  y0 = (y0 > 0.f) ? y0 : 0.f;
      float y1 = sv*(a1[v] + bv) + tv;  y1 = (y1 > 0.f) ? y1 : 0.f;
      ldsY3[m*32 + nl]      = y0;
      ldsY3[m*32 + 16 + nl] = y1;
    }
  }
  __syncthreads();

  // ---- max over 32 points, accumulate onto nf3d ----
  {
    int o = t;                       // 0..127
    float mv = ldsY3[o*32];
    #pragma unroll
    for (int n = 1; n < 32; ++n) {
      float x = ldsY3[o*32 + n];
      mv = (x > mv) ? x : mv;
    }
    int b = grp >> 10, s = grp & 1023;
    size_t idx = OUT_XYZ_ELEMS + ((size_t)b*128 + o)*SS + s;
    out[idx] += mv;                  // nf3d written by k_conv earlier on stream
  }
}

// ---------------------------------------------------------------------------
// Workspace layout (bytes, all 32B-aligned)
// ---------------------------------------------------------------------------
#define WS_FEAT  ((size_t)0)                  // 8192*32*96*2 = 50331648
#define WS_LOCS  ((size_t)50331648)           // 8192*32*3*4  =  3145728
#define WS_W0P   ((size_t)53477376)           // 81*4*512*2   =   331776
#define WS_W1P   ((size_t)53809152)           // 64*4*512*2   =   262144
#define WS_WAP   ((size_t)54071296)           // 3*4*512*2    =    12288
#define WS_WBP   ((size_t)54083584)           // 2*4*512*2    =     8192
#define WS_WCP   ((size_t)54091776)           // 2*8*512*2    =    16384

extern "C" void kernel_launch(void* const* d_in, const int* in_sizes, int n_in,
                              void* d_out, int out_size, void* d_ws, size_t ws_size,
                              hipStream_t stream) {
  const float* xyz    = (const float*)d_in[0];
  const float* points = (const float*)d_in[1];
  const int*   fps    = (const int*)d_in[2];
  const int*   ball   = (const int*)d_in[3];
  const float* w3d0 = (const float*)d_in[4];
  const float* b3d0 = (const float*)d_in[5];
  const float* s3d0 = (const float*)d_in[6];
  const float* t3d0 = (const float*)d_in[7];
  const float* w3d1 = (const float*)d_in[8];
  const float* b3d1 = (const float*)d_in[9];
  const float* s3d1 = (const float*)d_in[10];
  const float* t3d1 = (const float*)d_in[11];
  const float* w3d2 = (const float*)d_in[12];
  const float* b3d2 = (const float*)d_in[13];
  const float* s3d2 = (const float*)d_in[14];
  const float* t3d2 = (const float*)d_in[15];
  const float* w2d0 = (const float*)d_in[16];
  const float* b2d0 = (const float*)d_in[17];
  const float* s2d0 = (const float*)d_in[18];
  const float* t2d0 = (const float*)d_in[19];
  const float* w2d1 = (const float*)d_in[20];
  const float* b2d1 = (const float*)d_in[21];
  const float* s2d1 = (const float*)d_in[22];
  const float* t2d1 = (const float*)d_in[23];
  const float* w2d2 = (const float*)d_in[24];
  const float* b2d2 = (const float*)d_in[25];
  const float* s2d2 = (const float*)d_in[26];
  const float* t2d2 = (const float*)d_in[27];

  char* ws = (char*)d_ws;
  __bf16* feat = (__bf16*)(ws + WS_FEAT);
  float*  locs = (float*) (ws + WS_LOCS);
  __bf16* W0p  = (__bf16*)(ws + WS_W0P);
  __bf16* W1p  = (__bf16*)(ws + WS_W1P);
  __bf16* Wap  = (__bf16*)(ws + WS_WAP);
  __bf16* Wbp  = (__bf16*)(ws + WS_WBP);
  __bf16* Wcp  = (__bf16*)(ws + WS_WCP);
  float*  out  = (float*)d_out;

  k_gather<<<NGRP, 128, 0, stream>>>(xyz, points, fps, ball, out, feat, locs);

  k_packA<<<(K0STEPS*4*512 + 255)/256, 256, 0, stream>>>(w3d0, 1, 4, K0STEPS, 0,  W0p);
  k_packA<<<(K1STEPS*4*512 + 255)/256, 256, 0, stream>>>(w3d1, 2, 4, K1STEPS, 0,  W1p);
  k_packA<<<(3*4*512 + 255)/256, 256, 0, stream>>>(w2d0, 0, 4, 3, 67, Wap);
  k_packA<<<(2*4*512 + 255)/256, 256, 0, stream>>>(w2d1, 0, 4, 2, 64, Wbp);
  k_packA<<<(2*8*512 + 255)/256, 256, 0, stream>>>(w2d2, 0, 8, 2, 64, Wcp);

  k_conv<<<NGRP, 128, 0, stream>>>(feat, locs, W0p, W1p,
                                   b3d0, s3d0, t3d0,
                                   b3d1, s3d1, t3d1,
                                   w3d2, b3d2, s3d2, t3d2, out);

  k_mlp<<<NGRP, 128, 0, stream>>>(feat,
                                  Wap, b2d0, s2d0, t2d0,
                                  Wbp, b2d1, s2d1, t2d1,
                                  Wcp, b2d2, s2d2, t2d2, out);
}
